// MultiHeadAttention_4535485465129
// MI455X (gfx1250) — matrix operand order repaired and checked by reading, not compiled, not run on hardware
//
#include <hip/hip_runtime.h>

// ---------------------------------------------------------------------------
// MHA forward for B=4, S=2048, D=1024, H=16, DH=64 on gfx1250 (CDNA5).
// bf16 WMMA (v_wmma_f32_16x16x32_bf16), fp32 accum + fp32 softmax.
// Attention K/V tiles staged into LDS via the Tensor Data Mover (TDM),
// double-buffered with s_wait_tensorcnt.
// ---------------------------------------------------------------------------

#define BB   4
#define SS   2048
#define DD   1024
#define HH   16
#define DHH  64
#define BS   (BB * SS)        // 8192 rows
#define NQKV (3 * DD)         // 3072 projection columns

typedef __attribute__((ext_vector_type(16))) __bf16 v16bf;
typedef __attribute__((ext_vector_type(8)))  __bf16 v8bf;
typedef __attribute__((ext_vector_type(8)))  float  v8f;
typedef __attribute__((ext_vector_type(4)))  unsigned int tdm_v4u;
typedef __attribute__((ext_vector_type(4)))  int          tdm_v4i;
typedef __attribute__((ext_vector_type(8)))  int          tdm_v8i;

static __device__ inline v8f wmma_bf16(v16bf a, v16bf b, v8f c) {
  return __builtin_amdgcn_wmma_f32_16x16x32_bf16(
      false, a, false, b, (short)0, c, false, false);
}

static __device__ inline v16bf load16(const __bf16* p0, const __bf16* p1) {
  v8bf lo = *reinterpret_cast<const v8bf*>(p0);
  v8bf hi = *reinterpret_cast<const v8bf*>(p1);
  v16bf r;
#pragma unroll
  for (int i = 0; i < 8; ++i) { r[i] = lo[i]; r[8 + i] = hi[i]; }
  return r;
}

static __device__ inline v8f vzero8() {
  v8f z;
#pragma unroll
  for (int i = 0; i < 8; ++i) z[i] = 0.0f;
  return z;
}

// ---------------------------------------------------------------------------
// TDM: load a 2D tile (tile_d0 x tile_d1 elements of 2 bytes) from global to
// LDS (packed contiguously). D# bit layout per CDNA5 ISA ch.8.
// ---------------------------------------------------------------------------
static __device__ inline void tdm_load_2d(unsigned lds_off, const void* gptr,
                                          unsigned tile_d0, unsigned tile_d1,
                                          unsigned long long stride0_elems) {
  unsigned long long ga = (unsigned long long)(uintptr_t)gptr;
  tdm_v4u g0;
  g0[0] = 1u;                                          // count=1, user mode
  g0[1] = lds_off;                                     // lds_addr (bytes)
  g0[2] = (unsigned)(ga & 0xffffffffu);                // global_addr[31:0]
  g0[3] = (unsigned)((ga >> 32) & 0x01ffffffu)         // global_addr[56:32]
        | (2u << 30);                                  // type=2 ("image")
  tdm_v8i g1;
  // tensor dims == tile dims -> no OOB zero-fill inside the tile
  unsigned td0 = tile_d0, td1 = tile_d1;
  g1[0] = (int)(1u << 16);                             // data_size=1 (2 bytes)
  g1[1] = (int)((td0 & 0xffffu) << 16);                // tensor_dim0[15:0]
  g1[2] = (int)((td0 >> 16) & 0xffffu)                 // tensor_dim0[31:16]
        | (int)((td1 & 0xffffu) << 16);                // tensor_dim1[15:0]
  g1[3] = (int)((td1 >> 16) & 0xffffu)                 // tensor_dim1[31:16]
        | (int)((tile_d0 & 0xffffu) << 16);            // tile_dim0
  g1[4] = (int)(tile_d1 & 0xffffu);                    // tile_dim1 (tile_dim2=0)
  g1[5] = (int)(stride0_elems & 0xffffffffu);          // tensor_dim0_stride lo
  g1[6] = (int)((stride0_elems >> 32) & 0xffffu);      // tensor_dim0_stride hi
  g1[7] = 0;                                           // tensor_dim1_stride=0
  tdm_v4i gz = {0, 0, 0, 0};
#if defined(__clang_major__) && (__clang_major__ >= 23)
  tdm_v8i gz8 = {0, 0, 0, 0, 0, 0, 0, 0};
  __builtin_amdgcn_tensor_load_to_lds(g0, g1, gz, gz, gz8, 0);
#else
  __builtin_amdgcn_tensor_load_to_lds(g0, g1, gz, gz, 0);
#endif
}

// ---------------------------------------------------------------------------
// Prep kernels: fp32 -> bf16 conversion and weight transposition.
// ---------------------------------------------------------------------------
__global__ void cvt_x_kernel(const float* __restrict__ x, __bf16* __restrict__ xbf) {
  int idx = blockIdx.x * 256 + threadIdx.x;
  xbf[idx] = (__bf16)x[idx];
}

__global__ void prep_wqkv_kernel(const float* __restrict__ Wq,
                                 const float* __restrict__ Wk,
                                 const float* __restrict__ Wv,
                                 __bf16* __restrict__ wT) {
  int idx = blockIdx.x * 256 + threadIdx.x;            // NQKV*DD threads
  int d = idx & (DD - 1);
  int c = idx >> 10;
  int sec = c >> 10;
  int r = c & (DD - 1);
  int h = r >> 6;
  int e = r & 63;
  const float* W = (sec == 0) ? Wq : ((sec == 1) ? Wk : Wv);
  wT[idx] = (__bf16)W[((size_t)h * DD + d) * DHH + e];
}

__global__ void prep_wo_kernel(const float* __restrict__ Wo, __bf16* __restrict__ wT) {
  int idx = blockIdx.x * 256 + threadIdx.x;            // DD*DD threads
  int c = idx >> 10;
  int d = idx & (DD - 1);
  wT[idx] = (__bf16)Wo[(size_t)d * DD + c];
}

// ---------------------------------------------------------------------------
// QKV projection GEMM. One wave computes a 32x64 tile (2 M-tiles x 4 N-tiles:
// 12 b128 loads per 8 WMMAs). grid = (BS/32/8, NQKV/64), block = 256.
// Writes Q/K as [B,H,S,DH] bf16 and V transposed [B,H,DH,S] bf16.
// ---------------------------------------------------------------------------
__global__ void qkv_gemm_kernel(const __bf16* __restrict__ xbf,
                                const __bf16* __restrict__ wT,
                                const float* __restrict__ bq,
                                const float* __restrict__ bk,
                                const float* __restrict__ bv,
                                __bf16* __restrict__ qb,
                                __bf16* __restrict__ kb,
                                __bf16* __restrict__ vt) {
  const int wave = threadIdx.x >> 5;
  const int lane = threadIdx.x & 31;
  const int lr   = lane & 15;
  const int half = lane >> 4;

  const int mtile = blockIdx.x * 8 + wave;             // 0..255 (32 rows each)
  const int ngrp  = blockIdx.y;                        // 0..47
  const int row0  = mtile * 32;
  const int cbase = ngrp * 64;

  const __bf16* arow0 = xbf + (size_t)(row0 + lr) * DD;
  const __bf16* arow1 = xbf + (size_t)(row0 + 16 + lr) * DD;
  const __bf16* brow[4];
#pragma unroll
  for (int t = 0; t < 4; ++t) brow[t] = wT + (size_t)(cbase + t * 16 + lr) * DD;

  v8f acc[2][4];
#pragma unroll
  for (int m = 0; m < 2; ++m)
#pragma unroll
    for (int t = 0; t < 4; ++t) acc[m][t] = vzero8();

  for (int k0 = 0; k0 < DD; k0 += 32) {
    v16bf a0 = load16(arow0 + k0 + half * 8, arow0 + k0 + half * 8 + 16);
    v16bf a1 = load16(arow1 + k0 + half * 8, arow1 + k0 + half * 8 + 16);
#pragma unroll
    for (int t = 0; t < 4; ++t) {
      v16bf bfr = load16(brow[t] + k0 + half * 8, brow[t] + k0 + half * 8 + 16);
      acc[0][t] = wmma_bf16(a0, bfr, acc[0][t]);
      acc[1][t] = wmma_bf16(a1, bfr, acc[1][t]);
    }
  }

  const int sec  = cbase >> 10;                        // 0=Q 1=K 2=V
  const int csec = cbase & (DD - 1);
  const float* bias = (sec == 0) ? bq : ((sec == 1) ? bk : bv);

#pragma unroll
  for (int m = 0; m < 2; ++m) {
    const int sg0 = row0 + m * 16 + half * 8;          // first of 8 rows
    const int b   = sg0 >> 11;
    const int s0  = sg0 & (SS - 1);
#pragma unroll
    for (int t = 0; t < 4; ++t) {
      const int cc = csec + t * 16 + lr;
      const int h  = cc >> 6;
      const int dh = cc & 63;
      const float bi = bias[cc];
      if (sec < 2) {
        __bf16* dst = ((sec == 0) ? qb : kb) +
                      (((size_t)b * HH + h) * SS + s0) * DHH + dh;
#pragma unroll
        for (int r = 0; r < 8; ++r) dst[(size_t)r * DHH] = (__bf16)(acc[m][t][r] + bi);
      } else {
        v8bf pk;
#pragma unroll
        for (int r = 0; r < 8; ++r) pk[r] = (__bf16)(acc[m][t][r] + bi);
        __bf16* dst = vt + (((size_t)b * HH + h) * DHH + dh) * SS + s0;
        *reinterpret_cast<v8bf*>(dst) = pk;
      }
    }
  }
}

// ---------------------------------------------------------------------------
// Flash attention. All 8 waves of a block share one (b,h) head; K and V^T
// tiles for 32 keys are TDM-loaded into LDS (double buffered). Each wave owns
// a 16-query tile with online softmax. grid = (B*H, S/16/8), block = 256.
// ---------------------------------------------------------------------------
__global__ void attn_kernel(const __bf16* __restrict__ qb,
                            const __bf16* __restrict__ kb,
                            const __bf16* __restrict__ vt,
                            __bf16* __restrict__ zb) {
  __shared__ __align__(16) struct {
    __bf16 k[2][32][DHH];     // 32 keys x 64 dh, key-major        (2 x 4KB)
    __bf16 v[2][DHH][32];     // 64 dh rows x 32 keys (V^T tile)   (2 x 4KB)
    __bf16 p[8][16][32];      // per-wave P staging                (8KB)
  } sm;

  const int wave = threadIdx.x >> 5;
  const int lane = threadIdx.x & 31;
  const int lr   = lane & 15;
  const int half = lane >> 4;

  const int bh = blockIdx.x;                           // 0..63
  const int qt = blockIdx.y * 8 + wave;                // 0..127

  const __bf16* kbase = kb + (size_t)bh * SS * DHH;
  const __bf16* vbase = vt + (size_t)bh * DHH * SS;

  // Q fragments (two 32-wide K-dim steps over DH=64), kept in registers
  const __bf16* qrow = qb + ((size_t)bh * SS + qt * 16 + lr) * DHH;
  const v16bf Q0 = load16(qrow +  0 + half * 8, qrow +  0 + half * 8 + 16);
  const v16bf Q1 = load16(qrow + 32 + half * 8, qrow + 32 + half * 8 + 16);

  float mrow[8], lrow[8];
#pragma unroll
  for (int r = 0; r < 8; ++r) { mrow[r] = -1.0e30f; lrow[r] = 0.0f; }
  v8f acc[4];
#pragma unroll
  for (int t = 0; t < 4; ++t) acc[t] = vzero8();

  // Prologue: TDM-load first key block into buffer 0 (wave 0 only).
  if (wave == 0) {
    tdm_load_2d((unsigned)(uintptr_t)&sm.k[0][0][0], kbase, 32 * DHH, 1, 32 * DHH);
    tdm_load_2d((unsigned)(uintptr_t)&sm.v[0][0][0], vbase, 32, DHH, SS);
  }

  for (int kt = 0; kt < SS; kt += 32) {
    const int buf = (kt >> 5) & 1;
    if (wave == 0) {
      if (kt + 32 < SS) {
        // prefetch next block into the other buffer, then wait for current
        tdm_load_2d((unsigned)(uintptr_t)&sm.k[buf ^ 1][0][0],
                    kbase + (size_t)(kt + 32) * DHH, 32 * DHH, 1, 32 * DHH);
        tdm_load_2d((unsigned)(uintptr_t)&sm.v[buf ^ 1][0][0],
                    vbase + (kt + 32), 32, DHH, SS);
        __builtin_amdgcn_s_wait_tensorcnt(2);          // current buffer done
      } else {
        __builtin_amdgcn_s_wait_tensorcnt(0);
      }
    }
    __syncthreads();

    // ---- scores for 32 keys from LDS: 2 key tiles x 2 K-dim steps ---------
    v8f sc[2];
#pragma unroll
    for (int t = 0; t < 2; ++t) {
      sc[t] = vzero8();
      const __bf16* krow = &sm.k[buf][t * 16 + lr][0];
      v16bf K0 = load16(krow +  0 + half * 8, krow +  0 + half * 8 + 16);
      v16bf K1 = load16(krow + 32 + half * 8, krow + 32 + half * 8 + 16);
      sc[t] = wmma_bf16(Q0, K0, sc[t]);
      sc[t] = wmma_bf16(Q1, K1, sc[t]);
    }

    // ---- online softmax over the 32 new columns ---------------------------
#pragma unroll
    for (int r = 0; r < 8; ++r) {
      float a0 = sc[0][r] * 0.125f;                    // 1/sqrt(64)
      float a1 = sc[1][r] * 0.125f;
      float cm = fmaxf(a0, a1);
#pragma unroll
      for (int off = 1; off < 16; off <<= 1)
        cm = fmaxf(cm, __shfl_xor(cm, off, 32));
      const float nm = fmaxf(mrow[r], cm);
      const float p0 = __expf(a0 - nm);
      const float p1 = __expf(a1 - nm);
      float rs = p0 + p1;
#pragma unroll
      for (int off = 1; off < 16; off <<= 1)
        rs += __shfl_xor(rs, off, 32);
      const float alpha = __expf(mrow[r] - nm);
      lrow[r] = lrow[r] * alpha + rs;
      mrow[r] = nm;
#pragma unroll
      for (int t2 = 0; t2 < 4; ++t2) acc[t2][r] *= alpha;
      sm.p[wave][half * 8 + r][lr]      = (__bf16)p0;
      sm.p[wave][half * 8 + r][16 + lr] = (__bf16)p1;
    }
    __builtin_amdgcn_wave_barrier();
    v16bf P = load16(&sm.p[wave][lr][half * 8], &sm.p[wave][lr][16 + half * 8]);
    __builtin_amdgcn_wave_barrier();

    // ---- acc += P[16x32] * V[32x64] from LDS ------------------------------
#pragma unroll
    for (int t = 0; t < 4; ++t) {
      const __bf16* vrow = &sm.v[buf][t * 16 + lr][0];
      v16bf V = load16(vrow + half * 8, vrow + half * 8 + 16);
      acc[t] = wmma_bf16(P, V, acc[t]);
    }
    __syncthreads();   // all waves done with `buf` before it is overwritten
  }

  // ---- normalize and write Z (concat-head layout, bf16) -------------------
  const int b = bh >> 4;
  const int h = bh & 15;
  const size_t zrow0 = (size_t)b * SS + qt * 16 + half * 8;
#pragma unroll
  for (int t = 0; t < 4; ++t) {
    const int col = h * DHH + t * 16 + lr;
#pragma unroll
    for (int r = 0; r < 8; ++r)
      zb[(zrow0 + r) * DD + col] = (__bf16)(acc[t][r] / lrow[r]);
  }
}

// ---------------------------------------------------------------------------
// Output projection GEMM: 32x64 tile per wave. grid = (BS/32/8, DD/64).
// ---------------------------------------------------------------------------
__global__ void out_gemm_kernel(const __bf16* __restrict__ zb,
                                const __bf16* __restrict__ woT,
                                const float* __restrict__ bo,
                                float* __restrict__ out) {
  const int wave = threadIdx.x >> 5;
  const int lane = threadIdx.x & 31;
  const int lr   = lane & 15;
  const int half = lane >> 4;

  const int mtile = blockIdx.x * 8 + wave;
  const int ngrp  = blockIdx.y;
  const int row0  = mtile * 32;
  const int cbase = ngrp * 64;

  const __bf16* arow0 = zb + (size_t)(row0 + lr) * DD;
  const __bf16* arow1 = zb + (size_t)(row0 + 16 + lr) * DD;
  const __bf16* brow[4];
#pragma unroll
  for (int t = 0; t < 4; ++t) brow[t] = woT + (size_t)(cbase + t * 16 + lr) * DD;

  v8f acc[2][4];
#pragma unroll
  for (int m = 0; m < 2; ++m)
#pragma unroll
    for (int t = 0; t < 4; ++t) acc[m][t] = vzero8();

  for (int k0 = 0; k0 < DD; k0 += 32) {
    v16bf a0 = load16(arow0 + k0 + half * 8, arow0 + k0 + half * 8 + 16);
    v16bf a1 = load16(arow1 + k0 + half * 8, arow1 + k0 + half * 8 + 16);
#pragma unroll
    for (int t = 0; t < 4; ++t) {
      v16bf bfr = load16(brow[t] + k0 + half * 8, brow[t] + k0 + half * 8 + 16);
      acc[0][t] = wmma_bf16(a0, bfr, acc[0][t]);
      acc[1][t] = wmma_bf16(a1, bfr, acc[1][t]);
    }
  }

#pragma unroll
  for (int m = 0; m < 2; ++m) {
#pragma unroll
    for (int t = 0; t < 4; ++t) {
      const int cc = cbase + t * 16 + lr;
      const float bi = bo[cc];
#pragma unroll
      for (int r = 0; r < 8; ++r)
        out[(size_t)(row0 + m * 16 + half * 8 + r) * DD + cc] = acc[m][t][r] + bi;
    }
  }
}

// ---------------------------------------------------------------------------
// Launcher
// ---------------------------------------------------------------------------
extern "C" void kernel_launch(void* const* d_in, const int* in_sizes, int n_in,
                              void* d_out, int out_size, void* d_ws, size_t ws_size,
                              hipStream_t stream) {
  (void)in_sizes; (void)n_in; (void)out_size; (void)ws_size;

  const float* x  = (const float*)d_in[0];
  const float* Wq = (const float*)d_in[1];
  const float* bq = (const float*)d_in[2];
  const float* Wk = (const float*)d_in[3];
  const float* bk = (const float*)d_in[4];
  const float* Wv = (const float*)d_in[5];
  const float* bv = (const float*)d_in[6];
  const float* Wo = (const float*)d_in[7];
  const float* bo = (const float*)d_in[8];
  float* out = (float*)d_out;

  char* ws = (char*)d_ws;
  size_t off = 0;
  __bf16* xbf   = (__bf16*)(ws + off); off += (size_t)BS * DD * 2;      // 16 MiB
  __bf16* wqkvT = (__bf16*)(ws + off); off += (size_t)NQKV * DD * 2;    //  6 MiB
  __bf16* woT   = (__bf16*)(ws + off); off += (size_t)DD * DD * 2;      //  2 MiB
  __bf16* qb    = (__bf16*)(ws + off); off += (size_t)BS * DD * 2;      // 16 MiB
  __bf16* kb    = (__bf16*)(ws + off); off += (size_t)BS * DD * 2;      // 16 MiB
  __bf16* vt    = (__bf16*)(ws + off); off += (size_t)BS * DD * 2;      // 16 MiB
  __bf16* zb    = (__bf16*)(ws + off); off += (size_t)BS * DD * 2;      // 16 MiB

  cvt_x_kernel<<<(BS * DD) / 256, 256, 0, stream>>>(x, xbf);
  prep_wqkv_kernel<<<(NQKV * DD) / 256, 256, 0, stream>>>(Wq, Wk, Wv, wqkvT);
  prep_wo_kernel<<<(DD * DD) / 256, 256, 0, stream>>>(Wo, woT);

  qkv_gemm_kernel<<<dim3(BS / 32 / 8, NQKV / 64), 256, 0, stream>>>(
      xbf, wqkvT, bq, bk, bv, qb, kb, vt);

  attn_kernel<<<dim3(BB * HH, SS / 16 / 8), 256, 0, stream>>>(qb, kb, vt, zb);

  out_gemm_kernel<<<dim3(BS / 32 / 8, DD / 64), 256, 0, stream>>>(zb, woT, bo, out);
}
